// MH_GATv2_sagpool_GraphConv_exp_76390288327227
// MI455X (gfx1250) — matrix operand order repaired and checked by reading, not compiled, not run on hardware
//
#include <hip/hip_runtime.h>
#include <math.h>

#define NNODES   32000
#define NEDGES   640000
#define ETOT     (NEDGES + NNODES)   // edges + self loops
#define NPGRAPH  2000
#define NGRAPH   16
#define KKEEP    400
#define FIN      1038
#define KPAD     1056                // 33 * 32
#define HC       384                 // H*C
#define NOUT     768                 // 2*H*C  ([xl | xr])
#define CC       128
#define KDW      (KPAD / 2)          // dwords per bf16 row (528)

typedef __attribute__((ext_vector_type(16))) __bf16 v16bf;
typedef __attribute__((ext_vector_type(8)))  float  v8f;

union Frag { v16bf v; unsigned u[8]; };

__device__ __forceinline__ unsigned short f2bf(float f) {
  unsigned u = __float_as_uint(f);
  u += 0x7FFFu + ((u >> 16) & 1u);           // round-to-nearest-even
  return (unsigned short)(u >> 16);
}
__device__ __forceinline__ unsigned fflip(float f) {   // monotone float->uint
  unsigned u = __float_as_uint(f);
  return (u & 0x80000000u) ? ~u : (u | 0x80000000u);
}
__device__ __forceinline__ float funflip(unsigned v) {
  unsigned u = (v & 0x80000000u) ? (v ^ 0x80000000u) : ~v;
  return __uint_as_float(u);
}
__device__ __forceinline__ float lrelu(float x, float s) { return x > 0.f ? x : s * x; }

// ---- Wt[n][k] bf16, n in [0,768): n<384 -> Wl col n, else Wr col n-384; k pad 0
__global__ __launch_bounds__(256) void k_convert_w(
    const float* __restrict__ Wl, const float* __restrict__ Wr,
    unsigned short* __restrict__ Wt) {
  int idx = blockIdx.x * 256 + threadIdx.x;
  if (idx >= NOUT * KPAD) return;
  int n = idx / KPAD, k = idx % KPAD;
  float w = 0.f;
  if (k < FIN) w = (n < HC) ? Wl[k * HC + n] : Wr[k * HC + (n - HC)];
  Wt[idx] = f2bf(w);
}

// ---- xb[row][k] bf16, k pad 0
__global__ __launch_bounds__(256) void k_convert_x(
    const float* __restrict__ x, unsigned short* __restrict__ xb) {
  int idx = blockIdx.x * 256 + threadIdx.x;
  if (idx >= NNODES * KPAD) return;
  int r = idx / KPAD, k = idx % KPAD;
  xb[idx] = f2bf(k < FIN ? x[r * FIN + k] : 0.f);
}

// ---- GEMM: xlr[32000][768] = xb @ Wt^T (+bl on first 384 cols).
// Block = 8 waves, covers 2 M-tiles x 48 N-tiles. Wave owns 6 N-tiles x 2 M-tiles
// (12 f32 accumulators). Each B fragment feeds 2 WMMAs, each A fragment 6.
__global__ __launch_bounds__(256) void k_gemm(
    const unsigned short* __restrict__ xb, const unsigned short* __restrict__ Wt,
    const float* __restrict__ bl, float* __restrict__ xlr) {
  int wave = threadIdx.x >> 5, lane = threadIdx.x & 31;
  int hi = lane >> 4;        // half-wave select
  int lr = lane & 15;        // row/col within tile
  int tm0 = blockIdx.x * 2;  // first of 2 M-tiles
  int tnb = wave * 6;        // first of 6 N-tiles for this wave

  const unsigned* arow0 = (const unsigned*)xb + (size_t)(tm0 * 16 + lr) * KDW;
  const unsigned* arow1 = arow0 + 16 * KDW;
  const unsigned* brow[6];
#pragma unroll
  for (int t = 0; t < 6; ++t)
    brow[t] = (const unsigned*)Wt + (size_t)((tnb + t) * 16 + lr) * KDW;

  // per-lane dword offsets inside a 32-K step (16 dwords):
  // A frag: dwords [hi*4 .. hi*4+3] and [8+hi*4 .. 8+hi*4+3]
  // B frag: same dwords as A frag
  int a_lo = hi * 4, a_hi = 8 + hi * 4;
  int b_lo = hi * 8;

  v8f acc[2][6] = {};
#pragma unroll 1
  for (int kb = 0; kb < KPAD / 32; ++kb) {
    int base = kb * 16;
    Frag a0, a1;
#pragma unroll
    for (int v = 0; v < 4; ++v) {
      a0.u[v] = arow0[base + a_lo + v];
      a0.u[v + 4] = arow0[base + a_hi + v];
      a1.u[v] = arow1[base + a_lo + v];
      a1.u[v + 4] = arow1[base + a_hi + v];
    }
#pragma unroll
    for (int t = 0; t < 6; ++t) {
      Frag b;
#pragma unroll
      for (int v = 0; v < 8; ++v) b.u[v] = brow[t][base + (v < 4 ? a_lo + v : a_hi + v - 4)];
      acc[0][t] = __builtin_amdgcn_wmma_f32_16x16x32_bf16(false, a0.v, false, b.v,
                                                          (short)0, acc[0][t], false, false);
      acc[1][t] = __builtin_amdgcn_wmma_f32_16x16x32_bf16(false, a1.v, false, b.v,
                                                          (short)0, acc[1][t], false, false);
    }
  }
#pragma unroll
  for (int t = 0; t < 6; ++t) {
    int ng = (tnb + t) * 16 + lr;
    float bias = (ng < HC) ? bl[ng] : 0.f;   // hoisted: invariant over r and mt
#pragma unroll
    for (int mt = 0; mt < 2; ++mt) {
#pragma unroll
      for (int r = 0; r < 8; ++r) {
        int mg = (tm0 + mt) * 16 + r + (hi << 3);
        xlr[(size_t)mg * NOUT + ng] = acc[mt][t][r] + bias;
      }
    }
  }
}

// ---- mean of edge_attr (sums; divided by E at use site)
__global__ __launch_bounds__(256) void k_eamean(const float* __restrict__ ea,
                                                float* __restrict__ easum) {
  __shared__ float s0[256], s1[256];
  float a0 = 0.f, a1 = 0.f;
  for (int e = blockIdx.x * 256 + threadIdx.x; e < NEDGES; e += gridDim.x * 256) {
    a0 += ea[2 * e]; a1 += ea[2 * e + 1];
  }
  s0[threadIdx.x] = a0; s1[threadIdx.x] = a1; __syncthreads();
  for (int o = 128; o; o >>= 1) {
    if (threadIdx.x < o) { s0[threadIdx.x] += s0[threadIdx.x + o]; s1[threadIdx.x] += s1[threadIdx.x + o]; }
    __syncthreads();
  }
  if (threadIdx.x == 0) { atomicAdd(&easum[0], s0[0]); atomicAdd(&easum[1], s1[0]); }
}

// ---- Pass A: attention logits e[edge][h] + segment max (wave per edge)
__global__ __launch_bounds__(256) void k_passA(
    const int* __restrict__ ei0, const int* __restrict__ ei1,
    const float* __restrict__ ea, const float* __restrict__ easum,
    const float* __restrict__ xlr, const float* __restrict__ We,
    const float* __restrict__ att, float* __restrict__ ebuf,
    unsigned* __restrict__ emax) {
  int wid = (blockIdx.x * 256 + threadIdx.x) >> 5;
  int lane = threadIdx.x & 31;
  if (wid >= ETOT) return;
  int s, d; float a0, a1;
  if (wid < NEDGES) { s = ei0[wid]; d = ei1[wid]; a0 = ea[2 * wid]; a1 = ea[2 * wid + 1]; }
  else { s = d = wid - NEDGES; a0 = easum[0] * (1.f / NEDGES); a1 = easum[1] * (1.f / NEDGES); }
  const float* xls = xlr + (size_t)s * NOUT;
  const float* xrd = xlr + (size_t)d * NOUT + HC;
  float ph0 = 0.f, ph1 = 0.f, ph2 = 0.f;
#pragma unroll
  for (int q = 0; q < 4; ++q) {
    int c = lane + q * 32;
    {
      int hc = c;
      float m = xls[hc] + xrd[hc] + a0 * We[hc] + a1 * We[HC + hc];
      ph0 += lrelu(m, 0.2f) * att[hc];
    }
    {
      int hc = 128 + c;
      float m = xls[hc] + xrd[hc] + a0 * We[hc] + a1 * We[HC + hc];
      ph1 += lrelu(m, 0.2f) * att[hc];
    }
    {
      int hc = 256 + c;
      float m = xls[hc] + xrd[hc] + a0 * We[hc] + a1 * We[HC + hc];
      ph2 += lrelu(m, 0.2f) * att[hc];
    }
  }
  for (int o = 16; o; o >>= 1) {
    ph0 += __shfl_xor(ph0, o, 32);
    ph1 += __shfl_xor(ph1, o, 32);
    ph2 += __shfl_xor(ph2, o, 32);
  }
  if (lane == 0) {
    ebuf[wid * 3 + 0] = ph0; ebuf[wid * 3 + 1] = ph1; ebuf[wid * 3 + 2] = ph2;
    atomicMax(&emax[d * 3 + 0], fflip(ph0));
    atomicMax(&emax[d * 3 + 1], fflip(ph1));
    atomicMax(&emax[d * 3 + 2], fflip(ph2));
  }
}

// ---- Pass B: ex = exp(e - max), denom += ex (thread per edge; ex stored in place)
__global__ __launch_bounds__(256) void k_passB(
    const int* __restrict__ ei1, float* __restrict__ ebuf,
    const unsigned* __restrict__ emax, float* __restrict__ denom) {
  int e = blockIdx.x * 256 + threadIdx.x;
  if (e >= ETOT) return;
  int d = (e < NEDGES) ? ei1[e] : (e - NEDGES);
#pragma unroll
  for (int h = 0; h < 3; ++h) {
    float ex = expf(ebuf[e * 3 + h] - funflip(emax[d * 3 + h]));
    ebuf[e * 3 + h] = ex;
    atomicAdd(&denom[d * 3 + h], ex);
  }
}

// ---- Pass C: h[dst] += (1/3) * sum_h alpha_h * xl[src][h][:]  (wave per edge)
__global__ __launch_bounds__(256) void k_passC(
    const int* __restrict__ ei0, const int* __restrict__ ei1,
    const float* __restrict__ ebuf, const float* __restrict__ denom,
    const float* __restrict__ xlr, float* __restrict__ hout) {
  int wid = (blockIdx.x * 256 + threadIdx.x) >> 5;
  int lane = threadIdx.x & 31;
  if (wid >= ETOT) return;
  int s, d;
  if (wid < NEDGES) { s = ei0[wid]; d = ei1[wid]; }
  else { s = d = wid - NEDGES; }
  float a0 = ebuf[wid * 3 + 0] / denom[d * 3 + 0] * (1.f / 3.f);
  float a1 = ebuf[wid * 3 + 1] / denom[d * 3 + 1] * (1.f / 3.f);
  float a2 = ebuf[wid * 3 + 2] / denom[d * 3 + 2] * (1.f / 3.f);
  const float* xls = xlr + (size_t)s * NOUT;
  float* hp = hout + (size_t)d * CC;
#pragma unroll
  for (int q = 0; q < 4; ++q) {
    int c = lane + q * 32;
    float acc = a0 * xls[c] + a1 * xls[128 + c] + a2 * xls[256 + c];
    atomicAdd(&hp[c], acc);
  }
}

// ---- h = leaky(hpre + conv_bias, 0.1); accumulate per-channel sum/sumsq
__global__ __launch_bounds__(256) void k_hfinal(
    float* __restrict__ h, const float* __restrict__ cbias,
    float* __restrict__ bnsum, float* __restrict__ bnsq) {
  __shared__ float sa[256], sb[256];
  int c = threadIdx.x & 127, rsub = threadIdx.x >> 7;
  float s = 0.f, s2 = 0.f;
  for (int r = blockIdx.x * 2 + rsub; r < NNODES; r += gridDim.x * 2) {
    float v = h[(size_t)r * CC + c] + cbias[c];
    v = lrelu(v, 0.1f);
    h[(size_t)r * CC + c] = v;
    s += v; s2 += v * v;
  }
  sa[threadIdx.x] = s; sb[threadIdx.x] = s2; __syncthreads();
  if (rsub == 0) {
    atomicAdd(&bnsum[c], sa[c] + sa[c + 128]);
    atomicAdd(&bnsq[c], sb[c] + sb[c + 128]);
  }
}

// ---- batchnorm (batch stats)
__global__ __launch_bounds__(256) void k_bnnorm(
    float* __restrict__ h, const float* __restrict__ bnsum, const float* __restrict__ bnsq,
    const float* __restrict__ gamma, const float* __restrict__ beta) {
  int idx = blockIdx.x * 256 + threadIdx.x;
  if (idx >= NNODES * CC) return;
  int c = idx & 127;
  float mu = bnsum[c] * (1.f / NNODES);
  float var = bnsq[c] * (1.f / NNODES) - mu * mu;
  h[idx] = (h[idx] - mu) * rsqrtf(var + 1e-5f) * gamma[c] + beta[c];
}

// ---- per-node dots with gc_Wroot / gc_Wrel (wave per node)
__global__ __launch_bounds__(256) void k_scoredots(
    const float* __restrict__ h, const float* __restrict__ wroot,
    const float* __restrict__ wrel, float* __restrict__ sroot, float* __restrict__ srel) {
  int wid = (blockIdx.x * 256 + threadIdx.x) >> 5;
  int lane = threadIdx.x & 31;
  if (wid >= NNODES) return;
  float pr = 0.f, pl = 0.f;
#pragma unroll
  for (int q = 0; q < 4; ++q) {
    int c = lane + q * 32;
    float v = h[(size_t)wid * CC + c];
    pr += v * wroot[c]; pl += v * wrel[c];
  }
  for (int o = 16; o; o >>= 1) { pr += __shfl_xor(pr, o, 32); pl += __shfl_xor(pl, o, 32); }
  if (lane == 0) { sroot[wid] = pr; srel[wid] = pl; }
}

// ---- scalar scatter: sagg[dst] += srel[src]   (original edges only)
__global__ __launch_bounds__(256) void k_scatter(
    const int* __restrict__ ei0, const int* __restrict__ ei1,
    const float* __restrict__ srel, float* __restrict__ sagg) {
  int e = blockIdx.x * 256 + threadIdx.x;
  if (e >= NEDGES) return;
  atomicAdd(&sagg[ei1[e]], srel[ei0[e]]);
}

__global__ __launch_bounds__(256) void k_score(
    const float* __restrict__ sroot, const float* __restrict__ sagg,
    const float* __restrict__ gcb, float* __restrict__ score) {
  int i = blockIdx.x * 256 + threadIdx.x;
  if (i >= NNODES) return;
  score[i] = tanhf(sroot[i] + sagg[i] + gcb[0]);
}

// ---- per-graph top-K select + weighted mean pool (one block per graph)
__global__ __launch_bounds__(256) void k_topkpool(
    const float* __restrict__ score, const float* __restrict__ h,
    float* __restrict__ pooled) {
  __shared__ unsigned sflip[NPGRAPH];
  __shared__ float stanh[NPGRAPH];
  __shared__ unsigned char sel[NPGRAPH];
  __shared__ int cnt;
  int g = blockIdx.x, tid = threadIdx.x;
  for (int i = tid; i < NPGRAPH; i += 256) {
    float v = score[g * NPGRAPH + i];
    stanh[i] = v; sflip[i] = fflip(v);
  }
  __syncthreads();
  unsigned T = 0;
  for (int bit = 31; bit >= 0; --bit) {
    unsigned cand = T | (1u << bit);
    if (tid == 0) cnt = 0;
    __syncthreads();
    int lc = 0;
    for (int i = tid; i < NPGRAPH; i += 256) if (sflip[i] >= cand) ++lc;
    atomicAdd(&cnt, lc);
    __syncthreads();
    int cc = cnt;
    __syncthreads();
    if (cc >= KKEEP) T = cand;
  }
  // count strictly-greater, then take ties ascending by index
  if (tid == 0) cnt = 0;
  __syncthreads();
  {
    int lc = 0;
    for (int i = tid; i < NPGRAPH; i += 256) if (sflip[i] > T) ++lc;
    atomicAdd(&cnt, lc);
  }
  __syncthreads();
  if (tid == 0) {
    int need = KKEEP - cnt;
    for (int i = 0; i < NPGRAPH; ++i) {
      unsigned char f = 0;
      if (sflip[i] > T) f = 1;
      else if (sflip[i] == T && need > 0) { f = 1; --need; }
      sel[i] = f;
    }
  }
  __syncthreads();
  if (tid < CC) {
    float acc = 0.f;
    for (int i = 0; i < NPGRAPH; ++i)
      if (sel[i]) acc += stanh[i] * h[((size_t)g * NPGRAPH + i) * CC + tid];
    pooled[g * CC + tid] = acc * (1.f / KKEEP);
  }
}

// ---- MLP head (single block)
__global__ __launch_bounds__(256) void k_mlp(
    const float* __restrict__ p1, const float* __restrict__ p2,
    const float* __restrict__ W1, const float* __restrict__ b1,
    const float* __restrict__ W2, const float* __restrict__ b2,
    const float* __restrict__ Wo, const float* __restrict__ bo,
    float* __restrict__ out) {
  __shared__ float xc[NGRAPH * 256];
  __shared__ float t1[NGRAPH * 128];
  __shared__ float t2[NGRAPH * 64];
  int tid = threadIdx.x;
  for (int idx = tid; idx < NGRAPH * 256; idx += 256) {
    int b = idx >> 8, j = idx & 255;
    xc[idx] = (j < 128) ? p1[b * 128 + j] : p2[b * 128 + (j - 128)];
  }
  __syncthreads();
  for (int idx = tid; idx < NGRAPH * 128; idx += 256) {
    int b = idx >> 7, o = idx & 127;
    float a = b1[o];
    for (int j = 0; j < 256; ++j) a += xc[b * 256 + j] * W1[j * 128 + o];
    t1[idx] = lrelu(a, 0.1f);
  }
  __syncthreads();
  for (int idx = tid; idx < NGRAPH * 64; idx += 256) {
    int b = idx >> 6, o = idx & 63;
    float a = b2[o];
    for (int j = 0; j < 128; ++j) a += t1[b * 128 + j] * W2[j * 64 + o];
    t2[idx] = lrelu(a, 0.1f);
  }
  __syncthreads();
  if (tid < NGRAPH) {
    float a = bo[0];
    for (int j = 0; j < 64; ++j) a += t2[tid * 64 + j] * Wo[j];
    out[tid] = 1.f / (1.f + expf(-a));
  }
}

extern "C" void kernel_launch(void* const* d_in, const int* in_sizes, int n_in,
                              void* d_out, int out_size, void* d_ws, size_t ws_size,
                              hipStream_t stream) {
  const float* p1_x = (const float*)d_in[0];
  const float* p2_x = (const float*)d_in[1];
  const int* p1_ei = (const int*)d_in[2];
  const int* p2_ei = (const int*)d_in[3];
  const float* p1_ea = (const float*)d_in[4];
  const float* p2_ea = (const float*)d_in[5];
  const float* Wl = (const float*)d_in[6];
  const float* bl = (const float*)d_in[7];
  const float* Wr = (const float*)d_in[8];
  const float* We = (const float*)d_in[9];
  const float* att = (const float*)d_in[10];
  const float* conv_bias = (const float*)d_in[11];
  const float* gamma = (const float*)d_in[12];
  const float* beta = (const float*)d_in[13];
  const float* gc_Wrel = (const float*)d_in[14];
  const float* gc_Wroot = (const float*)d_in[15];
  const float* gc_b = (const float*)d_in[16];
  const float* W1 = (const float*)d_in[17];
  const float* b1 = (const float*)d_in[18];
  const float* W2 = (const float*)d_in[19];
  const float* b2 = (const float*)d_in[20];
  const float* Wo = (const float*)d_in[21];
  const float* bo = (const float*)d_in[22];
  float* out = (float*)d_out;

  // ---- workspace carve-up (256B aligned)
  char* base = (char*)d_ws;
  size_t off = 0;
  auto carve = [&](size_t bytes) -> char* {
    char* p = base + off;
    off = (off + bytes + 255) & ~(size_t)255;
    return p;
  };
  unsigned short* wt = (unsigned short*)carve((size_t)NOUT * KPAD * 2);
  unsigned short* xb = (unsigned short*)carve((size_t)NNODES * KPAD * 2);
  float* xlr = (float*)carve((size_t)NNODES * NOUT * 4);
  float* ebuf = (float*)carve((size_t)ETOT * 3 * 4);
  // zero-region (one memset per branch)
  size_t zbeg = off;
  float* easum = (float*)carve(2 * 4);
  unsigned* emax = (unsigned*)carve((size_t)NNODES * 3 * 4);
  float* denom = (float*)carve((size_t)NNODES * 3 * 4);
  float* hbuf = (float*)carve((size_t)NNODES * CC * 4);
  float* bnsum = (float*)carve(CC * 4);
  float* bnsq = (float*)carve(CC * 4);
  float* sagg = (float*)carve((size_t)NNODES * 4);
  size_t zlen = off - zbeg;
  float* sroot = (float*)carve((size_t)NNODES * 4);
  float* srel = (float*)carve((size_t)NNODES * 4);
  float* score = (float*)carve((size_t)NNODES * 4);
  float* pooled1 = (float*)carve((size_t)NGRAPH * CC * 4);
  float* pooled2 = (float*)carve((size_t)NGRAPH * CC * 4);
  (void)ws_size; (void)in_sizes; (void)n_in; (void)out_size;

  // weights: branch-independent
  k_convert_w<<<(NOUT * KPAD + 255) / 256, 256, 0, stream>>>(Wl, Wr, wt);

  const float* xs[2] = {p1_x, p2_x};
  const int* eis[2] = {p1_ei, p2_ei};
  const float* eas[2] = {p1_ea, p2_ea};
  float* pooleds[2] = {pooled1, pooled2};

  for (int b = 0; b < 2; ++b) {
    const int* ei0 = eis[b];
    const int* ei1 = eis[b] + NEDGES;
    hipMemsetAsync(base + zbeg, 0, zlen, stream);
    k_convert_x<<<(NNODES * KPAD + 255) / 256, 256, 0, stream>>>(xs[b], xb);
    k_gemm<<<NNODES / 32, 256, 0, stream>>>(xb, wt, bl, xlr);
    k_eamean<<<512, 256, 0, stream>>>(eas[b], easum);
    k_passA<<<ETOT / 8, 256, 0, stream>>>(ei0, ei1, eas[b], easum, xlr, We, att, ebuf, emax);
    k_passB<<<ETOT / 256, 256, 0, stream>>>(ei1, ebuf, emax, denom);
    k_passC<<<ETOT / 8, 256, 0, stream>>>(ei0, ei1, ebuf, denom, xlr, hbuf);
    k_hfinal<<<500, 256, 0, stream>>>(hbuf, conv_bias, bnsum, bnsq);
    k_bnnorm<<<(NNODES * CC) / 256, 256, 0, stream>>>(hbuf, bnsum, bnsq, gamma, beta);
    k_scoredots<<<NNODES / 8, 256, 0, stream>>>(hbuf, gc_Wroot, gc_Wrel, sroot, srel);
    k_scatter<<<NEDGES / 256, 256, 0, stream>>>(ei0, ei1, srel, sagg);
    k_score<<<(NNODES + 255) / 256, 256, 0, stream>>>(sroot, sagg, gc_b, score);
    k_topkpool<<<NGRAPH, 256, 0, stream>>>(score, hbuf, pooleds[b]);
  }
  k_mlp<<<1, 256, 0, stream>>>(pooled1, pooled2, W1, b1, W2, b2, Wo, bo, out);
}
